// DenseVariational_62577673502970
// MI455X (gfx1250) — matrix operand order repaired and checked by reading, not compiled, not run on hardware
//
#include <hip/hip_runtime.h>

#define S_DIM   32
#define B_DIM   256
#define IN_DIM  1024
#define OUT_DIM 1024
#define NTILES  8                    // 16-wide N-subtiles per workgroup (128-wide slab)
#define LDS_STRIDE 12                // uints per lane fragment (8 used + 4 pad)
#define TILE_U  (32 * LDS_STRIDE)    // uints per N-subtile
#define BUF_U   (NTILES * TILE_U)    // uints per k-step buffer (3072 = 12 KB)

typedef __attribute__((ext_vector_type(16))) __bf16 v16bf;
typedef __attribute__((ext_vector_type(2)))  __bf16 v2bf;
typedef __attribute__((ext_vector_type(8)))  float  v8f;

union Frag16 { unsigned int u[8]; v16bf v; };

__device__ __forceinline__ float softplus_f(float x) {
    // stable: max(x,0) + log1p(exp(-|x|))
    float e = __expf(-__builtin_fabsf(x));
    return fmaxf(x, 0.0f) + __logf(1.0f + e);
}

__device__ __forceinline__ unsigned int pack_bf16(float a, float b) {
#if __has_builtin(__builtin_amdgcn_cvt_pk_bf16_f32)
    union { v2bf h; unsigned int u; } cv;
    cv.h = __builtin_amdgcn_cvt_pk_bf16_f32(a, b);
    return cv.u;
#else
    unsigned int ua = __float_as_uint(a);
    ua += 0x7FFFu + ((ua >> 16) & 1u);
    unsigned int ub = __float_as_uint(b);
    ub += 0x7FFFu + ((ub >> 16) & 1u);
    return (ua >> 16) | (ub & 0xFFFF0000u);
#endif
}

// one-shot: sigma = softplus(rho) over the (OUT,IN) weight matrix
__global__ __launch_bounds__(256)
void softplus_precompute(const float* __restrict__ rho, float* __restrict__ sig) {
    const int i = (blockIdx.x * 256 + threadIdx.x) * 4;
    float4 r = *reinterpret_cast<const float4*>(rho + i);
    float4 o;
    o.x = softplus_f(r.x); o.y = softplus_f(r.y);
    o.z = softplus_f(r.z); o.w = softplus_f(r.w);
    *reinterpret_cast<float4*>(sig + i) = o;
}

template<bool PRE>
__global__ __launch_bounds__(256)
void dv_wmma(const float* __restrict__ input,
             const float* __restrict__ wmu,
             const float* __restrict__ wsr,   // sigma if PRE, else rho
             const float* __restrict__ bmu,
             const float* __restrict__ brho,
             const float* __restrict__ epsw,
             const float* __restrict__ epsb,
             float* __restrict__ out)
{
    __shared__ unsigned int ldsB[4 * BUF_U];   // 48 KB, 4 buffers

    const int tid  = threadIdx.x;
    const int lane = tid & 31;
    const int wave = tid >> 5;              // 0..7
    const int s    = blockIdx.y;            // sample
    const int o0   = blockIdx.x * (NTILES * 16);
    const int mbase = wave * 32;            // 2 M-tiles per wave

    // ---- cooperative weight-slab builder: 128 o x 32 k, 16 k-elems per thread
    const int bo  = tid >> 1;               // 0..127: o within slab
    const int gp  = (tid & 1) * 2;          // octet pair: k = gp*8 .. gp*8+15
    const int bnt = bo >> 4;                // N-subtile
    const int bn  = bo & 15;
    const int bL  = ((gp >> 1) << 4) + bn;  // dest WMMA B lane (k>=16 -> +16)
    const int bslot = bnt * TILE_U + bL * LDS_STRIDE;   // store q0 @bslot, q1 @bslot+4
    const long wrow = (long)(o0 + bo) * IN_DIM + (gp << 3);
    const float4* mu_p = reinterpret_cast<const float4*>(wmu + wrow);
    const float4* sr_p = reinterpret_cast<const float4*>(wsr + wrow);
    const float4* ep_p = reinterpret_cast<const float4*>(
        epsw + (long)s * (OUT_DIM * IN_DIM) + wrow);

    // ---- A-operand mapping (16-bit A 16x32 layout)
    const int arow  = mbase + (lane & 15);
    const int khalf = lane >> 4;            // elems 0..15: K=khalf*16+.
    const float4* in0 = reinterpret_cast<const float4*>(
        input + (long)(s * B_DIM + arow) * IN_DIM);
    const float4* in1 = in0 + (16 * IN_DIM) / 4;

    v8f acc[2][NTILES];
    #pragma unroll
    for (int m = 0; m < 2; ++m)
        #pragma unroll
        for (int t = 0; t < NTILES; ++t) acc[m][t] = (v8f){};

    // build one 128x32 bf16 weight slab (this thread's 16 elements) into buffer `buf`
    auto buildSlab = [&](int k0, int buf) {
        const int f = k0 >> 2;
        float4 m0 = mu_p[f], m1 = mu_p[f + 1], m2 = mu_p[f + 2], m3 = mu_p[f + 3];
        float4 s0 = sr_p[f], s1 = sr_p[f + 1], s2 = sr_p[f + 2], s3 = sr_p[f + 3];
        float4 e0 = ep_p[f], e1 = ep_p[f + 1], e2 = ep_p[f + 2], e3 = ep_p[f + 3];
        if (!PRE) {
            s0.x = softplus_f(s0.x); s0.y = softplus_f(s0.y);
            s0.z = softplus_f(s0.z); s0.w = softplus_f(s0.w);
            s1.x = softplus_f(s1.x); s1.y = softplus_f(s1.y);
            s1.z = softplus_f(s1.z); s1.w = softplus_f(s1.w);
            s2.x = softplus_f(s2.x); s2.y = softplus_f(s2.y);
            s2.z = softplus_f(s2.z); s2.w = softplus_f(s2.w);
            s3.x = softplus_f(s3.x); s3.y = softplus_f(s3.y);
            s3.z = softplus_f(s3.z); s3.w = softplus_f(s3.w);
        }
        uint4 q0, q1;
        q0.x = pack_bf16(fmaf(s0.x, e0.x, m0.x), fmaf(s0.y, e0.y, m0.y));
        q0.y = pack_bf16(fmaf(s0.z, e0.z, m0.z), fmaf(s0.w, e0.w, m0.w));
        q0.z = pack_bf16(fmaf(s1.x, e1.x, m1.x), fmaf(s1.y, e1.y, m1.y));
        q0.w = pack_bf16(fmaf(s1.z, e1.z, m1.z), fmaf(s1.w, e1.w, m1.w));
        q1.x = pack_bf16(fmaf(s2.x, e2.x, m2.x), fmaf(s2.y, e2.y, m2.y));
        q1.y = pack_bf16(fmaf(s2.z, e2.z, m2.z), fmaf(s2.w, e2.w, m2.w));
        q1.z = pack_bf16(fmaf(s3.x, e3.x, m3.x), fmaf(s3.y, e3.y, m3.y));
        q1.w = pack_bf16(fmaf(s3.z, e3.z, m3.z), fmaf(s3.w, e3.w, m3.w));
        unsigned int* dst = &ldsB[buf * BUF_U + bslot];
        *reinterpret_cast<uint4*>(dst)     = q0;
        *reinterpret_cast<uint4*>(dst + 4) = q1;
    };

    // one 32-K compute step: 2 M-tiles x 8 N-subtiles = 16 WMMAs
    auto compute = [&](int k0, int buf) {
        const int c = (k0 + khalf * 16) >> 2;
        Frag16 fa0, fa1;
        {
            float4 a0 = in0[c], a1 = in0[c + 1], a2 = in0[c + 2], a3 = in0[c + 3];
            fa0.u[0] = pack_bf16(a0.x, a0.y); fa0.u[1] = pack_bf16(a0.z, a0.w);
            fa0.u[2] = pack_bf16(a1.x, a1.y); fa0.u[3] = pack_bf16(a1.z, a1.w);
            fa0.u[4] = pack_bf16(a2.x, a2.y); fa0.u[5] = pack_bf16(a2.z, a2.w);
            fa0.u[6] = pack_bf16(a3.x, a3.y); fa0.u[7] = pack_bf16(a3.z, a3.w);
        }
        {
            float4 a0 = in1[c], a1 = in1[c + 1], a2 = in1[c + 2], a3 = in1[c + 3];
            fa1.u[0] = pack_bf16(a0.x, a0.y); fa1.u[1] = pack_bf16(a0.z, a0.w);
            fa1.u[2] = pack_bf16(a1.x, a1.y); fa1.u[3] = pack_bf16(a1.z, a1.w);
            fa1.u[4] = pack_bf16(a2.x, a2.y); fa1.u[5] = pack_bf16(a2.z, a2.w);
            fa1.u[6] = pack_bf16(a3.x, a3.y); fa1.u[7] = pack_bf16(a3.z, a3.w);
        }
        #pragma unroll
        for (int t = 0; t < NTILES; ++t) {
            Frag16 bf;
            const uint4* bp = reinterpret_cast<const uint4*>(
                &ldsB[buf * BUF_U + t * TILE_U + lane * LDS_STRIDE]);
            uint4 q0 = bp[0], q1 = bp[1];
            bf.u[0] = q0.x; bf.u[1] = q0.y; bf.u[2] = q0.z; bf.u[3] = q0.w;
            bf.u[4] = q1.x; bf.u[5] = q1.y; bf.u[6] = q1.z; bf.u[7] = q1.w;
            acc[0][t] = __builtin_amdgcn_wmma_f32_16x16x32_bf16(
                false, fa0.v, false, bf.v, (short)0, acc[0][t], false, false);
            acc[1][t] = __builtin_amdgcn_wmma_f32_16x16x32_bf16(
                false, fa1.v, false, bf.v, (short)0, acc[1][t], false, false);
        }
    };

    // 64-K outer iterations, one barrier each, buffer pairs ping-pong {0,1}/{2,3}
    for (int k0 = 0; k0 < IN_DIM; k0 += 64) {
        const int bb = ((k0 >> 6) & 1) * 2;
        buildSlab(k0,      bb);
        buildSlab(k0 + 32, bb + 1);
        __syncthreads();
        compute(k0,      bb);
        compute(k0 + 32, bb + 1);
    }

    // -------- epilogue: reparameterized bias + store (C/D 16x16 f32 layout)
    const int n = lane & 15;
    const int rbase = mbase + khalf * 8;    // lanes>=16 hold M = v+8
    #pragma unroll
    for (int t = 0; t < NTILES; ++t) {
        const int o = o0 + t * 16 + n;
        const float bias = fmaf(softplus_f(brho[o]), epsb[s * OUT_DIM + o], bmu[o]);
        float* op0 = out + (long)(s * B_DIM + rbase) * OUT_DIM + o;
        float* op1 = op0 + 16 * OUT_DIM;
        #pragma unroll
        for (int v = 0; v < 8; ++v) {
            op0[(long)v * OUT_DIM] = acc[0][t][v] + bias;
            op1[(long)v * OUT_DIM] = acc[1][t][v] + bias;
        }
    }
}

extern "C" void kernel_launch(void* const* d_in, const int* in_sizes, int n_in,
                              void* d_out, int out_size, void* d_ws, size_t ws_size,
                              hipStream_t stream) {
    const float* input = (const float*)d_in[0];
    const float* wmu   = (const float*)d_in[1];
    const float* wrho  = (const float*)d_in[2];
    const float* bmu   = (const float*)d_in[3];
    const float* brho  = (const float*)d_in[4];
    const float* epsw  = (const float*)d_in[5];
    const float* epsb  = (const float*)d_in[6];
    float* out = (float*)d_out;

    dim3 grid(OUT_DIM / (16 * NTILES), S_DIM);   // (8, 32)

    if (ws_size >= (size_t)OUT_DIM * IN_DIM * sizeof(float)) {
        float* sig = (float*)d_ws;
        softplus_precompute<<<(OUT_DIM * IN_DIM) / (256 * 4), 256, 0, stream>>>(wrho, sig);
        dv_wmma<true><<<grid, 256, 0, stream>>>(
            input, wmu, sig, bmu, brho, epsw, epsb, out);
    } else {
        dv_wmma<false><<<grid, 256, 0, stream>>>(
            input, wmu, wrho, bmu, brho, epsw, epsb, out);
    }
}
